// EnhanceSelfAttention_23742579212429
// MI455X (gfx1250) — matrix operand order repaired and checked by reading, not compiled, not run on hardware
//
#include <hip/hip_runtime.h>
#include <hip/hip_bf16.h>

// ---------------------------------------------------------------------------
// EnhanceSelfAttention for MI455X (gfx1250, wave32)
//   WMMA f32_16x16x32_f16 everywhere + TDM (tensor_load_to_lds) K/V staging
//   B=2, S=2048, C=1024, H=16, D=64
// ---------------------------------------------------------------------------

typedef __attribute__((ext_vector_type(16))) _Float16 v16h;
typedef __attribute__((ext_vector_type(8)))  float    v8f;
typedef __attribute__((ext_vector_type(4)))  unsigned int tdm_v4u;
typedef __attribute__((ext_vector_type(8)))  int          tdm_v8i;
typedef __attribute__((ext_vector_type(4)))  int          tdm_v4i;

#define B_  2
#define S_  2048
#define C_  1024
#define H_  16
#define D_  64

__device__ __forceinline__ int lane_id() { return (int)(threadIdx.x & 31u); }

__device__ __forceinline__ v8f wmma_f16(v16h a, v16h b, v8f c) {
  // D = A(16x32 f16) * B(32x16 f16) + C(16x16 f32)
  return __builtin_amdgcn_wmma_f32_16x16x32_f16(false, a, false, b,
                                                (short)0, c, false, false);
}

// ---- A fragment: row-major [16 x ld] tile, lane=M, interleaved K (ISA 7.12.2)
template <typename T>
__device__ __forceinline__ v16h load_a_frag(const T* src, int ld) {
  const int l  = lane_id();
  const int m  = l & 15;
  const int kb = (l < 16) ? 0 : 8;
  v16h a;
#pragma unroll
  for (int j = 0; j < 8; ++j) {
    const int k = (j < 4) ? (kb + 2 * j) : (16 + kb + 2 * (j - 4));
    a[2 * j]     = (_Float16)src[m * ld + k];
    a[2 * j + 1] = (_Float16)src[m * ld + k + 1];
  }
  return a;
}

__device__ __forceinline__ v16h load_a_frag_scaled(const _Float16* src, int ld,
                                                   float s) {
  const int l  = lane_id();
  const int m  = l & 15;
  const int kb = (l < 16) ? 0 : 8;
  v16h a;
#pragma unroll
  for (int j = 0; j < 8; ++j) {
    const int k = (j < 4) ? (kb + 2 * j) : (16 + kb + 2 * (j - 4));
    a[2 * j]     = (_Float16)((float)src[m * ld + k] * s);
    a[2 * j + 1] = (_Float16)((float)src[m * ld + k + 1] * s);
  }
  return a;
}

// ---- B fragment: row-major [32 x ld] (K x N) tile, lane=N col, interleaved K
template <typename T>
__device__ __forceinline__ v16h load_b_frag(const T* src, int ld) {
  const int l  = lane_id();
  const int n  = l & 15;
  const int kb = (l < 16) ? 0 : 8;
  v16h b;
#pragma unroll
  for (int j = 0; j < 16; ++j) {
    const int k = (j < 8) ? (kb + j) : (16 + kb + (j - 8));
    b[j] = (_Float16)src[k * ld + n];
  }
  return b;
}

// ---- B fragment of K^T for scores: B[d][key] = Ktile[key][doff + d], lane=key
__device__ __forceinline__ v16h load_kt_frag(const _Float16* p, int doff) {
  const int l  = lane_id();
  const int n  = l & 15;
  const int kb = (l < 16) ? 0 : 8;
  v16h b;
#pragma unroll
  for (int j = 0; j < 16; ++j) {
    const int k = (j < 8) ? (kb + j) : (16 + kb + (j - 8));
    b[j] = p[n * D_ + doff + k];
  }
  return b;
}

// 16-lane (half-wave) reductions; xor masks < 16 stay within each half
__device__ __forceinline__ float rmax16(float v) {
#pragma unroll
  for (int o = 1; o < 16; o <<= 1) v = fmaxf(v, __shfl_xor(v, o, 32));
  return v;
}
__device__ __forceinline__ float rsum16(float v) {
#pragma unroll
  for (int o = 1; o < 16; o <<= 1) v += __shfl_xor(v, o, 32);
  return v;
}

// ---------------------------------------------------------------------------
// TDM: 1-D contiguous f16 copy global -> LDS via Tensor DMA descriptor (D#)
// per CDNA5 ISA. n_elems fits tile_dim0 (16-bit).
__device__ __forceinline__ void tdm_load_1d(unsigned int lds_byte_addr,
                                            const void* gptr,
                                            unsigned int n_elems) {
  const unsigned long long ga = (unsigned long long)gptr;
  tdm_v4u g0;
  g0[0] = 1u;                                            // count=1 (valid)
  g0[1] = lds_byte_addr;                                 // lds_addr
  g0[2] = (unsigned int)(ga & 0xffffffffull);            // global_addr[31:0]
  g0[3] = (unsigned int)((ga >> 32) & 0x01ffffffull)     // global_addr[56:32]
        | (2u << 30);                                    // type=2 ("image")
  tdm_v8i g1;
  g1[0] = (int)(1u << 16);                               // data_size=2B
  g1[1] = (int)((n_elems & 0xffffu) << 16);              // tensor_dim0 lo16
  g1[2] = (int)(((n_elems >> 16) & 0xffffu) | (1u << 16)); // td0 hi | tdim1=1
  g1[3] = (int)((n_elems & 0xffffu) << 16);              // tile_dim0
  g1[4] = 0;                                             // tile_dim1/2 unused
  g1[5] = (int)n_elems;                                  // dim0_stride lo32
  g1[6] = 0;
  g1[7] = 0;
  const tdm_v4i z4 = {0, 0, 0, 0};
#if __has_include(<hip/amd_detail/amd_gfx1250_TDM.h>)
  const tdm_v8i z8 = {0, 0, 0, 0, 0, 0, 0, 0};
  __builtin_amdgcn_tensor_load_to_lds(g0, g1, z4, z4, z8, 0);   // clang-23 form
#else
  __builtin_amdgcn_tensor_load_to_lds(g0, g1, z4, z4, 0);       // ROCm 7.2 form
#endif
}

// ---------------------------------------------------------------------------
__global__ void cvt_kernel(const float* __restrict__ src,
                           _Float16* __restrict__ dst, int n) {
  int i = blockIdx.x * blockDim.x + threadIdx.x;
  if (i < n) dst[i] = (_Float16)src[i];
}

// ---------------------------------------------------------------------------
// qkv = x @ w_qkv (M=4096,K=1024,N=3072) + q/v bias + RoPE, scatter [B,H,S,D].
// Each wave: one 16x64 strip (A fragment reused 4x). 4 waves/block.
__global__ void qkv_rope_kernel(const _Float16* __restrict__ xh,
                                const _Float16* __restrict__ wh,
                                const float* __restrict__ q_bias,
                                const float* __restrict__ v_bias,
                                _Float16* __restrict__ qh,
                                _Float16* __restrict__ kh,
                                _Float16* __restrict__ vh) {
  const int w   = threadIdx.x >> 5;
  const int t   = blockIdx.x * 4 + w;      // 0 .. 12287
  const int mt  = t / 48;                  // 256 M-tiles
  const int nt4 = t % 48;                  // 48 strips of 64 cols
  const int n0  = nt4 * 64;

  const _Float16* arow = xh + (size_t)(mt * 16) * C_;
  const _Float16* bcol = wh + n0;

  v8f acc[4] = {v8f{}, v8f{}, v8f{}, v8f{}};
#pragma unroll 2
  for (int kk = 0; kk < 32; ++kk) {
    const v16h a = load_a_frag(arow + kk * 32, C_);
    if (kk + 1 < 32) __builtin_prefetch(arow + (kk + 1) * 32, 0, 1);
    const _Float16* bp = bcol + (size_t)(kk * 32) * (3 * C_);
    acc[0] = wmma_f16(a, load_b_frag(bp + 0, 3 * C_), acc[0]);
    acc[1] = wmma_f16(a, load_b_frag(bp + 16, 3 * C_), acc[1]);
    acc[2] = wmma_f16(a, load_b_frag(bp + 32, 3 * C_), acc[2]);
    acc[3] = wmma_f16(a, load_b_frag(bp + 48, 3 * C_), acc[3]);
  }

  const int l      = lane_id();
  const int n_loc  = l & 15;
  const int rowoff = (l >> 4) * 8;
  const int mat    = n0 >> 10;             // 0=q 1=k 2=v (64-strip = 1 head)
  const int c0     = n0 & (C_ - 1);
  const int h      = c0 >> 6;
  _Float16* dst = (mat == 0) ? qh : ((mat == 1) ? kh : vh);

#pragma unroll
  for (int dt = 0; dt < 4; ++dt) {
    const int d = dt * 16 + n_loc;
    const int c = c0 + d;
    float bias = 0.f;
    if (mat == 0) bias = q_bias[c];
    else if (mat == 2) bias = v_bias[c];
    const float inv_freq = __powf(1.0e4f, -(float)(d >> 1) / 32.0f);
#pragma unroll
    for (int r = 0; r < 8; ++r) {
      const int row = mt * 16 + r + rowoff;   // = b*S + s
      const int bb  = row >> 11;
      const int s   = row & (S_ - 1);
      float val = acc[dt][r] + bias;
      if (mat != 2) {                         // RoPE on q,k (uniform branch)
        const float ang = (float)s * inv_freq;
        const float cs = __cosf(ang), sn = __sinf(ang);
        const float other = __shfl_xor(val, 1, 32);     // pair partner d^1
        val = (d & 1) ? (val * cs + other * sn)
                      : (val * cs - other * sn);
      }
      dst[(((size_t)(bb * H_ + h)) * S_ + s) * D_ + d] = (_Float16)val;
    }
  }
}

// ---------------------------------------------------------------------------
// Causal flash attention. Block = (b,h, 64-row q block); 4 waves own one
// 16-row q tile each. 32-key K/V chunks staged into LDS by the TDM and
// shared by all 4 waves. P transposed through per-wave LDS scratch.
__global__ void attn_kernel(const _Float16* __restrict__ qh,
                            const _Float16* __restrict__ kh,
                            const _Float16* __restrict__ vh,
                            _Float16* __restrict__ oh) {
  __shared__ _Float16 ldsK[32 * D_];       // 4 KB
  __shared__ _Float16 ldsV[32 * D_];       // 4 KB
  __shared__ _Float16 lds_p[4][16 * 32];   // 4 KB (per-wave P scratch)

  const int w  = threadIdx.x >> 5;
  const int l  = lane_id();
  const int qb = blockIdx.x & 31;          // 32 q-blocks of 64 rows
  const int bh = blockIdx.x >> 5;          // b*H + h
  const int bb = bh >> 4, h = bh & 15;
  const int qt = qb * 4 + w;               // this wave's 16-row q tile

  const _Float16* Q  = qh + (size_t)bh * S_ * D_ + (size_t)qt * 16 * D_;
  const _Float16* Kp = kh + (size_t)bh * S_ * D_;
  const _Float16* Vp = vh + (size_t)bh * S_ * D_;

  const v16h qa0 = load_a_frag_scaled(Q, D_, 0.125f);       // d 0..31
  const v16h qa1 = load_a_frag_scaled(Q + 32, D_, 0.125f);  // d 32..63

  v8f acc0 = {}, acc1 = {}, acc2 = {}, acc3 = {};
  float rmax[8], rsum[8];
#pragma unroll
  for (int r = 0; r < 8; ++r) { rmax[r] = -3.0e38f; rsum[r] = 0.f; }

  const int rowoff = (l >> 4) * 8;
  const int n_loc  = l & 15;
  const int qbase  = qt * 16;
  const int nsteps = qb * 2 + 2;           // covers keys 0 .. qb*64+63

  const unsigned int ldsK_off = (unsigned int)(unsigned long long)&ldsK[0];
  const unsigned int ldsV_off = (unsigned int)(unsigned long long)&ldsV[0];

  for (int j = 0; j < nsteps; ++j) {
    const int k0 = j * 32;
    __syncthreads();                       // prior-step LDS reads done
    if (w == 0) {                          // TDM: async DMA 4KB K + 4KB V
      tdm_load_1d(ldsK_off, (const void*)(Kp + (size_t)k0 * D_), 32 * D_);
      tdm_load_1d(ldsV_off, (const void*)(Vp + (size_t)k0 * D_), 32 * D_);
      __builtin_amdgcn_s_wait_tensorcnt(0);
    }
    __syncthreads();                       // publish staged K/V

    v8f s0 = {}, s1 = {};
    s0 = wmma_f16(qa0, load_kt_frag(ldsK, 0), s0);
    s0 = wmma_f16(qa1, load_kt_frag(ldsK, 32), s0);            // keys +0..15
    s1 = wmma_f16(qa0, load_kt_frag(ldsK + 16 * D_, 0), s1);
    s1 = wmma_f16(qa1, load_kt_frag(ldsK + 16 * D_, 32), s1);  // keys +16..31

#pragma unroll
    for (int r = 0; r < 8; ++r) {
      const int qg = qbase + r + rowoff;
      float s0r = (k0 + n_loc      > qg) ? -1.0e9f : s0[r];
      float s1r = (k0 + 16 + n_loc > qg) ? -1.0e9f : s1[r];
      const float rm   = rmax16(fmaxf(s0r, s1r));
      const float mnew = fmaxf(rmax[r], rm);
      const float corr = __expf(rmax[r] - mnew);
      rmax[r] = mnew;
      const float p0 = __expf(s0r - mnew);
      const float p1 = __expf(s1r - mnew);
      rsum[r] = rsum[r] * corr + rsum16(p0 + p1);
      acc0[r] *= corr; acc1[r] *= corr; acc2[r] *= corr; acc3[r] *= corr;
      const int m = r + rowoff;
      lds_p[w][m * 32 + n_loc]      = (_Float16)p0;   // D-layout -> row-major
      lds_p[w][m * 32 + 16 + n_loc] = (_Float16)p1;
    }

    // Re-read P as A fragment (16 q x 32 keys); per-wave LDS, DS in-order
    const v16h pa = load_a_frag(&lds_p[w][0], 32);
    acc0 = wmma_f16(pa, load_b_frag(ldsV + 0, D_), acc0);
    acc1 = wmma_f16(pa, load_b_frag(ldsV + 16, D_), acc1);
    acc2 = wmma_f16(pa, load_b_frag(ldsV + 32, D_), acc2);
    acc3 = wmma_f16(pa, load_b_frag(ldsV + 48, D_), acc3);
  }

#pragma unroll
  for (int r = 0; r < 8; ++r) {
    const float inv = 1.0f / rsum[r];
    const int spos = qbase + r + rowoff;
    const size_t base = ((size_t)(bb * S_ + spos)) * C_ + h * 64;
    oh[base + 0  + n_loc] = (_Float16)(acc0[r] * inv);
    oh[base + 16 + n_loc] = (_Float16)(acc1[r] * inv);
    oh[base + 32 + n_loc] = (_Float16)(acc2[r] * inv);
    oh[base + 48 + n_loc] = (_Float16)(acc3[r] * inv);
  }
}

// ---------------------------------------------------------------------------
// out = oh @ w_out (M=4096,K=1024,N=1024), fp32 result; 16x64 strip per wave.
__global__ void proj_kernel(const _Float16* __restrict__ ah,
                            const _Float16* __restrict__ wh,
                            float* __restrict__ out) {
  const int w   = threadIdx.x >> 5;
  const int t   = blockIdx.x * 4 + w;      // 0 .. 4095
  const int mt  = t >> 4;                  // 256 M-tiles
  const int nt4 = t & 15;                  // 16 strips of 64 cols
  const int n0  = nt4 * 64;

  const _Float16* arow = ah + (size_t)(mt * 16) * C_;
  const _Float16* bcol = wh + n0;

  v8f acc[4] = {v8f{}, v8f{}, v8f{}, v8f{}};
#pragma unroll 2
  for (int kk = 0; kk < 32; ++kk) {
    const v16h a = load_a_frag(arow + kk * 32, C_);
    if (kk + 1 < 32) __builtin_prefetch(arow + (kk + 1) * 32, 0, 1);
    const _Float16* bp = bcol + (size_t)(kk * 32) * C_;
    acc[0] = wmma_f16(a, load_b_frag(bp + 0, C_), acc[0]);
    acc[1] = wmma_f16(a, load_b_frag(bp + 16, C_), acc[1]);
    acc[2] = wmma_f16(a, load_b_frag(bp + 32, C_), acc[2]);
    acc[3] = wmma_f16(a, load_b_frag(bp + 48, C_), acc[3]);
  }

  const int l = lane_id();
  const int n_loc = l & 15;
  const int rowoff = (l >> 4) * 8;
#pragma unroll
  for (int dt = 0; dt < 4; ++dt)
#pragma unroll
    for (int r = 0; r < 8; ++r)
      out[(size_t)(mt * 16 + r + rowoff) * C_ + n0 + dt * 16 + n_loc] =
          acc[dt][r];
}

// ---------------------------------------------------------------------------
extern "C" void kernel_launch(void* const* d_in, const int* in_sizes, int n_in,
                              void* d_out, int out_size, void* d_ws,
                              size_t ws_size, hipStream_t stream) {
  const float* x      = (const float*)d_in[0];  // [B,S,C]
  const float* w_qkv  = (const float*)d_in[1];  // [C,3C]
  const float* q_bias = (const float*)d_in[2];  // [C]
  const float* v_bias = (const float*)d_in[3];  // [C]
  const float* w_out  = (const float*)d_in[4];  // [C,C]
  float* out = (float*)d_out;

  const size_t NX    = (size_t)B_ * S_ * C_;      // 4194304
  const size_t NWQKV = (size_t)C_ * 3 * C_;       // 3145728
  const size_t NWOUT = (size_t)C_ * C_;           // 1048576
  const size_t NHD   = (size_t)B_ * H_ * S_ * D_; // 4194304

  char* ws = (char*)d_ws;
  size_t off = 0;
  auto alloc = [&](size_t bytes) {
    void* p = ws + off;
    off += (bytes + 255) & ~(size_t)255;
    return p;
  };
  _Float16* xh    = (_Float16*)alloc(NX * 2);
  _Float16* wqkvh = (_Float16*)alloc(NWQKV * 2);
  _Float16* wouth = (_Float16*)alloc(NWOUT * 2);
  _Float16* qhp   = (_Float16*)alloc(NHD * 2);
  _Float16* khp   = (_Float16*)alloc(NHD * 2);
  _Float16* vhp   = (_Float16*)alloc(NHD * 2);
  _Float16* ohp   = (_Float16*)alloc(NX * 2);
  (void)ws_size; (void)n_in; (void)in_sizes; (void)out_size;

  cvt_kernel<<<(int)(NX / 256), 256, 0, stream>>>(x, xh, (int)NX);
  cvt_kernel<<<(int)(NWQKV / 256), 256, 0, stream>>>(w_qkv, wqkvh, (int)NWQKV);
  cvt_kernel<<<(int)(NWOUT / 256), 256, 0, stream>>>(w_out, wouth, (int)NWOUT);

  // 256 M-tiles * 48 strips / 4 waves per 128-thread block
  qkv_rope_kernel<<<3072, 128, 0, stream>>>(xh, wqkvh, q_bias, v_bias,
                                            qhp, khp, vhp);
  // 32 (b,h) * 32 q-blocks
  attn_kernel<<<1024, 128, 0, stream>>>(qhp, khp, vhp, ohp);
  // 256 M-tiles * 16 strips / 4 waves per block
  proj_kernel<<<1024, 128, 0, stream>>>(ohp, wouth, out);
}
